// WaveSubsystem_43379169689842
// MI455X (gfx1250) — matrix operand order repaired and checked by reading, not compiled, not run on hardware
//
#include <hip/hip_runtime.h>
#include <math.h>

typedef float v8f  __attribute__((ext_vector_type(8)));
typedef __bf16 v2bf  __attribute__((ext_vector_type(2)));
typedef __bf16 v8bf  __attribute__((ext_vector_type(8)));
typedef __bf16 v16bf __attribute__((ext_vector_type(16)));

#define B_SZ 4096
#define DW   2048
#define DC   1024
#define DP   2048
#define RANK 512
#define KB   64           // K per LDS stage (2 wmma k-steps)
#define RS   72           // LDS row stride in bf16 elems (144 B: 16B-aligned, bank-spread)
#define TILE (128 * RS)   // one tile buffer (elems)

// ---- split two floats into packed-hi / packed-lo bf16 pairs (RNE, hw cvt) ----
__device__ __forceinline__ void split2(float a, float b, unsigned& h, unsigned& l) {
  v2bf hv; hv.x = (__bf16)a; hv.y = (__bf16)b;
  float ra = a - (float)hv.x;
  float rb = b - (float)hv.y;
  v2bf lv; lv.x = (__bf16)ra; lv.y = (__bf16)rb;
  h = __builtin_bit_cast(unsigned, hv);
  l = __builtin_bit_cast(unsigned, lv);
}

// ---- stage a 128-row x 64-K tile from row-major [rows, ld] f32 into hi/lo bf16 LDS ----
__device__ __forceinline__ void stage_rowmajor(unsigned short* __restrict__ hi,
                                               unsigned short* __restrict__ lo,
                                               const float* __restrict__ src,
                                               int row0, int k0, int ld, int tid) {
  int c  = tid & 15;   // float4 along K (16*4 = 64)
  int r0 = tid >> 4;   // 0..15
#pragma unroll
  for (int i = 0; i < 8; ++i) {
    int r = r0 + i * 16;
    const float* p = src + (size_t)(row0 + r) * ld + k0 + c * 4;
    float4 v = *(const float4*)p;
    uint2 h, l;
    split2(v.x, v.y, h.x, l.x);
    split2(v.z, v.w, h.y, l.y);
    *(uint2*)(hi + r * RS + c * 4) = h;
    *(uint2*)(lo + r * RS + c * 4) = l;
  }
}

// ---- stage a 64-K x 128-N tile from natural [K, N] f32 source into hi/lo [n][k] LDS ----
__device__ __forceinline__ void stage_kmajor(unsigned short* __restrict__ hi,
                                             unsigned short* __restrict__ lo,
                                             const float* __restrict__ src,
                                             int k0, int col0, int ld, int tid) {
  int n  = tid & 127;  // coalesced across lanes
  int kg = tid >> 7;   // 0..1
#pragma unroll
  for (int i = 0; i < 8; ++i) {
    int k4 = (kg + i * 2) * 4;  // 0,8,..,56 | 4,12,..,60
    const float* p = src + (size_t)(k0 + k4) * ld + col0 + n;
    float a  = p[0];
    float b  = p[(size_t)ld];
    float c2 = p[2 * (size_t)ld];
    float d  = p[3 * (size_t)ld];
    uint2 h, l;
    split2(a, b, h.x, l.x);
    split2(c2, d, h.y, l.y);
    *(uint2*)(hi + n * RS + k4) = h;
    *(uint2*)(lo + n * RS + k4) = l;
  }
}

// ---- WMMA fragment loads (ISA 16-bit layouts), rows K-contiguous in LDS ----
__device__ __forceinline__ v16bf fragA(const unsigned short* t, int row, int ks, int h) {
  // lane half h: VGPR0-3 = K h*8..h*8+7 ; VGPR4-7 = K 16+h*8..
  const __bf16* p = (const __bf16*)(t + row * RS + ks * 32);
  v8bf a = *(const v8bf*)(p + h * 8);
  v8bf b = *(const v8bf*)(p + 16 + h * 8);
  return __builtin_shufflevector(a, b, 0,1,2,3,4,5,6,7,8,9,10,11,12,13,14,15);
}
__device__ __forceinline__ v16bf fragB(const unsigned short* t, int row, int ks, int h) {
  // lane half h: VGPR0-3 = K h*8..h*8+7 ; VGPR4-7 = K 16+h*8..
  const __bf16* p = (const __bf16*)(t + row * RS + ks * 32);
  v8bf a = *(const v8bf*)(p + h * 8);
  v8bf b = *(const v8bf*)(p + 16 + h * 8);
  return __builtin_shufflevector(a, b, 0,1,2,3,4,5,6,7,8,9,10,11,12,13,14,15);
}

// ---- one KB=64 chunk: 2 k-steps x (2x4 tiles) x 3 split-WMMAs ----
__device__ __forceinline__ void mma_chunk(const unsigned short* Ahi, const unsigned short* Alo,
                                          const unsigned short* Bhi, const unsigned short* Blo,
                                          int wm0, int wn0, int lane, v8f (&acc)[2][4]) {
  int lm = lane & 15, h = lane >> 4;
#pragma unroll
  for (int ks = 0; ks < KB / 32; ++ks) {
    v16bf ah[2], al[2];
#pragma unroll
    for (int mt = 0; mt < 2; ++mt) {
      ah[mt] = fragA(Ahi, wm0 + mt * 16 + lm, ks, h);
      al[mt] = fragA(Alo, wm0 + mt * 16 + lm, ks, h);
    }
#pragma unroll
    for (int nt = 0; nt < 4; ++nt) {
      v16bf bh = fragB(Bhi, wn0 + nt * 16 + lm, ks, h);
      v16bf bl = fragB(Blo, wn0 + nt * 16 + lm, ks, h);
#pragma unroll
      for (int mt = 0; mt < 2; ++mt) {
        acc[mt][nt] = __builtin_amdgcn_wmma_f32_16x16x32_bf16(
            false, ah[mt], false, bh, (short)0, acc[mt][nt], false, false);
        acc[mt][nt] = __builtin_amdgcn_wmma_f32_16x16x32_bf16(
            false, ah[mt], false, bl, (short)0, acc[mt][nt], false, false);
        acc[mt][nt] = __builtin_amdgcn_wmma_f32_16x16x32_bf16(
            false, al[mt], false, bh, (short)0, acc[mt][nt], false, false);
      }
    }
  }
}

__device__ __forceinline__ void zero_acc(v8f (&acc)[2][4]) {
  v8f z = {0.f, 0.f, 0.f, 0.f, 0.f, 0.f, 0.f, 0.f};
#pragma unroll
  for (int mt = 0; mt < 2; ++mt)
#pragma unroll
    for (int nt = 0; nt < 4; ++nt) acc[mt][nt] = z;
}

// ---- double-buffered pipelined GEMM K-loop.
// BMODE 0: B natural [K,N] (stage_kmajor); BMODE 1: B is [N,K] row-major (stage_rowmajor)
template <int BMODE>
__device__ __forceinline__ void gemm_phase(unsigned short* Ahi, unsigned short* Alo,
                                           unsigned short* Bhi, unsigned short* Blo,
                                           const float* Asrc, int am0, int ald,
                                           const float* Bsrc, int bn0, int bld,
                                           int K, int wm0, int wn0, int lane, int tid,
                                           v8f (&acc)[2][4]) {
  __syncthreads();  // previous phase may still be reading these buffers
  stage_rowmajor(Ahi, Alo, Asrc, am0, 0, ald, tid);
  if (BMODE == 0) stage_kmajor(Bhi, Blo, Bsrc, 0, bn0, bld, tid);
  else            stage_rowmajor(Bhi, Blo, Bsrc, bn0, 0, bld, tid);
  int nb = K / KB;
  for (int i = 0; i < nb; ++i) {
    int cur = (i & 1) * TILE, nxt = TILE - cur;
    __syncthreads();  // stage of 'cur' visible; mma on 'nxt' from last iter done
    if (i + 1 < nb) {
      int k0 = (i + 1) * KB;
      stage_rowmajor(Ahi + nxt, Alo + nxt, Asrc, am0, k0, ald, tid);
      if (BMODE == 0) stage_kmajor(Bhi + nxt, Blo + nxt, Bsrc, k0, bn0, bld, tid);
      else            stage_rowmajor(Bhi + nxt, Blo + nxt, Bsrc, bn0, k0, bld, tid);
    }
    mma_chunk(Ahi + cur, Alo + cur, Bhi + cur, Blo + cur, wm0, wn0, lane, acc);
  }
}

// ================= Kernel 1: T = w_prev @ A_V  ([4096,2048]@[2048,512]) =================
__global__ __launch_bounds__(256) void t_gemm_kernel(const float* __restrict__ wprev,
                                                     const float* __restrict__ Av,
                                                     float* __restrict__ T) {
  __shared__ unsigned short Ahi[2 * TILE], Alo[2 * TILE], Bhi[2 * TILE], Blo[2 * TILE];
  int tid = threadIdx.x;
  int wave = tid >> 5, lane = tid & 31;
  int wm0 = (wave & 3) * 32, wn0 = (wave >> 2) * 64;
  int bm0 = blockIdx.y * 128, bn0 = blockIdx.x * 128;

  v8f acc[2][4];
  zero_acc(acc);
  gemm_phase<0>(Ahi, Alo, Bhi, Blo, wprev, bm0, DW, Av, bn0, RANK, DW,
                wm0, wn0, lane, tid, acc);
#pragma unroll
  for (int mt = 0; mt < 2; ++mt) {
    int mbase = bm0 + wm0 + mt * 16 + ((lane >> 4) << 3);
#pragma unroll
    for (int nt = 0; nt < 4; ++nt) {
      int col = bn0 + wn0 + nt * 16 + (lane & 15);
#pragma unroll
      for (int r = 0; r < 8; ++r)
        T[(size_t)(mbase + r) * RANK + col] = acc[mt][nt][r];
    }
  }
}

// ====== Kernel 2: fused modulation / Aw / Bt / output over [128x128] tiles of out ======
__global__ __launch_bounds__(256) void fused_kernel(
    const float* __restrict__ T, const float* __restrict__ c_t,
    const float* __restrict__ z_t, const float* __restrict__ wprev,
    const float* __restrict__ A_diag, const float* __restrict__ A_U,
    const float* __restrict__ W_mod, const float* __restrict__ b_mod,
    const float* __restrict__ W_B, const float* __restrict__ b_B,
    float* __restrict__ out) {
  __shared__ unsigned short Ahi[2 * TILE], Alo[2 * TILE], Bhi[2 * TILE], Blo[2 * TILE];
  int tid = threadIdx.x;
  int wave = tid >> 5, lane = tid & 31;
  int wm0 = (wave & 3) * 32, wn0 = (wave >> 2) * 64;
  int lm = lane & 15;
  int bm0 = blockIdx.y * 128, bn0 = blockIdx.x * 128;

  v8f acc[2][4];
  v8f keep[2][4];

  // ---------------- Phase 1: modulation = tanh(c_t @ W_mod + b_mod) ----------------
  zero_acc(acc);
  gemm_phase<0>(Ahi, Alo, Bhi, Blo, c_t, bm0, DC, W_mod, bn0, DW, DC,
                wm0, wn0, lane, tid, acc);
#pragma unroll
  for (int nt = 0; nt < 4; ++nt) {
    int col = bn0 + wn0 + nt * 16 + lm;
    float bm = b_mod[col];
#pragma unroll
    for (int mt = 0; mt < 2; ++mt)
#pragma unroll
      for (int r = 0; r < 8; ++r)
        keep[mt][nt][r] = tanhf(acc[mt][nt][r] + bm);
  }

  // ---------------- Phase 2: keep *= a_diag*w_prev + T @ A_U^T ----------------
  zero_acc(acc);
  // A_U is [DW, RANK] row-major; consumed as A_U^T -> K-contiguous both ends
  gemm_phase<1>(Ahi, Alo, Bhi, Blo, T, bm0, RANK, A_U, bn0, RANK, RANK,
                wm0, wn0, lane, tid, acc);
#pragma unroll
  for (int nt = 0; nt < 4; ++nt) {
    int col = bn0 + wn0 + nt * 16 + lm;
    float ad = 0.9f * tanhf(A_diag[col]);
#pragma unroll
    for (int mt = 0; mt < 2; ++mt) {
      int mbase = bm0 + wm0 + mt * 16 + ((lane >> 4) << 3);
#pragma unroll
      for (int r = 0; r < 8; ++r) {
        float wp = wprev[(size_t)(mbase + r) * DW + col];
        keep[mt][nt][r] *= (ad * wp + acc[mt][nt][r]);
      }
    }
  }

  // ---------------- Phase 3: Bt = [c_t | z_t] @ W_B + b_B; out = keep + Bt ----------------
  zero_acc(acc);
  gemm_phase<0>(Ahi, Alo, Bhi, Blo, c_t, bm0, DC, W_B, bn0, DW, DC,
                wm0, wn0, lane, tid, acc);
  gemm_phase<0>(Ahi, Alo, Bhi, Blo, z_t, bm0, DP, W_B + (size_t)DC * DW, bn0, DW, DP,
                wm0, wn0, lane, tid, acc);
#pragma unroll
  for (int nt = 0; nt < 4; ++nt) {
    int col = bn0 + wn0 + nt * 16 + lm;
    float bb = b_B[col];
#pragma unroll
    for (int mt = 0; mt < 2; ++mt) {
      int mbase = bm0 + wm0 + mt * 16 + ((lane >> 4) << 3);
#pragma unroll
      for (int r = 0; r < 8; ++r)
        out[(size_t)(mbase + r) * DW + col] = keep[mt][nt][r] + acc[mt][nt][r] + bb;
    }
  }
}

extern "C" void kernel_launch(void* const* d_in, const int* in_sizes, int n_in,
                              void* d_out, int out_size, void* d_ws, size_t ws_size,
                              hipStream_t stream) {
  (void)in_sizes; (void)n_in; (void)out_size; (void)ws_size;
  const float* w_prev = (const float*)d_in[0];
  const float* z_t    = (const float*)d_in[1];
  const float* c_t    = (const float*)d_in[2];
  const float* A_diag = (const float*)d_in[3];
  const float* A_U    = (const float*)d_in[4];
  const float* A_V    = (const float*)d_in[5];
  const float* W_mod  = (const float*)d_in[6];
  const float* b_mod  = (const float*)d_in[7];
  const float* W_B    = (const float*)d_in[8];
  const float* b_B    = (const float*)d_in[9];
  float* out = (float*)d_out;
  float* T   = (float*)d_ws;   // [4096, 512] fp32 = 8 MB scratch

  dim3 g1(RANK / 128, B_SZ / 128);  // 4 x 32
  dim3 g2(DW / 128, B_SZ / 128);    // 16 x 32
  t_gemm_kernel<<<g1, 256, 0, stream>>>(w_prev, A_V, T);
  fused_kernel<<<g2, 256, 0, stream>>>(T, c_t, z_t, w_prev, A_diag, A_U,
                                       W_mod, b_mod, W_B, b_B, out);
}
